// Model_6648609374812
// MI455X (gfx1250) — matrix operand order repaired and checked by reading, not compiled, not run on hardware
//
#include <hip/hip_runtime.h>
#include <hip/hip_bf16.h>

// ---------------- model dims ----------------
#define Bn 16
#define Ln 400
#define Tn 100
#define En 128
#define Hn 256
#define Vn 50000
#define G4 1024   // 4*H
#define H2 512    // 2*H
#define H3 768    // 3*H
#define NT (Tn-1) // 99 decoder steps

typedef __bf16 bf16_t;
typedef bf16_t v16bf __attribute__((ext_vector_type(16)));
typedef float  v8f   __attribute__((ext_vector_type(8)));

#define WMMA_BF16(a,b,c) \
  __builtin_amdgcn_wmma_f32_16x16x32_bf16(false,(a),false,(b),(short)0,(c),false,false)

__device__ __forceinline__ bf16_t f2bf(float f) {
  union { float f; unsigned u; } x; x.f = f;
  unsigned u = x.u;
  unsigned r = u + 0x7FFFu + ((u >> 16) & 1u);  // round-to-nearest-even
  unsigned short h = (unsigned short)(r >> 16);
  union { unsigned short s; bf16_t b; } y; y.s = h; return y.b;
}

__device__ __forceinline__ float sigm(float x) { return 1.0f / (1.0f + __expf(-x)); }

// A-fragment: 16x32 bf16 tile of row-major X[16][lda], rows = M (batch).
// lane<16: row=lane,  K = k0 + 0..15
// lane>=16: row=lane-16, K = k0 + 16..31
__device__ __forceinline__ v16bf load_a16(const bf16_t* X, int lda, int k0, int lane) {
  int row = lane & 15;
  int khi = (lane & 16) ? 16 : 0;
  const bf16_t* p = X + (size_t)row * lda + k0 + khi;
  v16bf a;
#pragma unroll
  for (int j = 0; j < 16; ++j) a[j] = p[j];
  return a;
}

// B-fragment for D = A * W^T : B[k][n] = W[(n0+n)*ldw + k0+k], W row-major [N][K].
// lane<16: n=lane,  K = k0 + 0..15 ; lane>=16: n=lane-16, K = k0 + 16..31
__device__ __forceinline__ v16bf load_bt16(const bf16_t* W, int ldw, int n0, int k0, int lane) {
  int n = lane & 15;
  int kh = (lane & 16) ? 16 : 0;
  const bf16_t* p = W + (size_t)(n0 + n) * ldw + (k0 + kh);
  v16bf b;
#pragma unroll
  for (int j = 0; j < 16; ++j) b[j] = p[j];
  return b;
}

// ---------------- utility kernels ----------------
__global__ __launch_bounds__(256) void f32_to_bf16_k(const float* __restrict__ src,
                                                     bf16_t* __restrict__ dst, size_t n) {
  size_t i = (size_t)blockIdx.x * 256 + threadIdx.x;
  if (i < n) dst[i] = f2bf(src[i]);
}

__global__ __launch_bounds__(256) void gather_xs_k(const int* __restrict__ inputs,
                                                   const float* __restrict__ embed,
                                                   bf16_t* __restrict__ xs) { // [L][B][E]
  size_t idx = (size_t)blockIdx.x * 256 + threadIdx.x;
  if (idx >= (size_t)Ln * Bn * En) return;
  int e = (int)(idx & (En - 1));
  size_t r = idx >> 7;            // l*Bn + b
  int b = (int)(r & (Bn - 1));
  int l = (int)(r >> 4);
  int tok = inputs[b * Ln + l];
  xs[idx] = f2bf(embed[(size_t)tok * En + e]);
}

__global__ __launch_bounds__(256) void init_state_k(float* __restrict__ cov,
                                                    float* __restrict__ c_g,
                                                    bf16_t* __restrict__ h_g,
                                                    float* __restrict__ covloss) {
  int i = blockIdx.x * 256 + threadIdx.x;
  if (i < Bn * Ln) cov[i] = 0.0f;
  if (i < Bn * Hn) { c_g[i] = 0.0f; h_g[i] = f2bf(0.0f); }
  if (i == 0) covloss[0] = 0.0f;
}

// ---------------- encoder: persistent bi-LSTM scan ----------------
// grid = 2 (block 0 forward, block 1 backward), block = 256 (8 waves).
// Wave w owns gate columns [w*32, w*32+32); acc tile pt = p*2+t covers gate p
// (i,f,g,o) columns [w*32 + t*16, +16), so the LSTM elementwise runs entirely
// in the WMMA D-register layout.  An asm memory clobber at the top of each
// step stops LICM from hoisting the (loop-invariant) weight B-fragments out
// of the scan -- they must stream from WGP$/L2 each step instead of being
// spilled to scratch.
__global__ __launch_bounds__(256)
void lstm_encoder_scan_k(const bf16_t* __restrict__ xs,
                         const bf16_t* __restrict__ WihF, const bf16_t* __restrict__ WhhF,
                         const float* __restrict__ bF,
                         const bf16_t* __restrict__ WihB, const bf16_t* __restrict__ WhhB,
                         const float* __restrict__ bB,
                         float* __restrict__ hF, float* __restrict__ hB) { // [L][B][H]
  __shared__ bf16_t h_bf[Bn * Hn];
  int tid = threadIdx.x, lane = tid & 31, w = tid >> 5;
  bool rev = (blockIdx.x == 1);
  const bf16_t* Wih = rev ? WihB : WihF;
  const bf16_t* Whh = rev ? WhhB : WhhF;
  const float*  bias = rev ? bB : bF;
  float* hout = rev ? hB : hF;

  for (int i = tid; i < Bn * Hn; i += 256) h_bf[i] = f2bf(0.0f);
  __syncthreads();

  int n = lane & 15;
  int rbase = (lane & 16) ? 8 : 0;

  // bias lives in registers across the whole scan (loaded before the clobber)
  float bval[8];
#pragma unroll
  for (int pt = 0; pt < 8; ++pt)
    bval[pt] = bias[(pt >> 1) * Hn + w * 32 + (pt & 1) * 16 + n];

  float creg[2][8];
#pragma unroll
  for (int t = 0; t < 2; ++t)
#pragma unroll
    for (int r = 0; r < 8; ++r) creg[t][r] = 0.0f;

  for (int s = 0; s < Ln; ++s) {
    asm volatile("" ::: "memory");   // block LICM / keep weight loads per-step
    int l = rev ? (Ln - 1 - s) : s;
    const bf16_t* X = xs + (size_t)l * Bn * En;

    v8f acc[8];
#pragma unroll
    for (int pt = 0; pt < 8; ++pt) {
      v8f a;
#pragma unroll
      for (int r = 0; r < 8; ++r) a[r] = bval[pt];
      acc[pt] = a;
    }
    // x @ Wih^T   (K = 128): per kk load A + 8 B-frags, then 8 WMMAs
#pragma unroll
    for (int kk = 0; kk < 4; ++kk) {
      v16bf af = load_a16(X, En, kk * 32, lane);
      v16bf bw[8];
#pragma unroll
      for (int pt = 0; pt < 8; ++pt)
        bw[pt] = load_bt16(Wih, En, (pt >> 1) * Hn + w * 32 + (pt & 1) * 16, kk * 32, lane);
#pragma unroll
      for (int pt = 0; pt < 8; ++pt)
        acc[pt] = WMMA_BF16(af, bw[pt], acc[pt]);
    }
    // h @ Whh^T   (K = 256)
#pragma unroll
    for (int kk = 0; kk < 8; ++kk) {
      v16bf ah = load_a16(h_bf, Hn, kk * 32, lane);
      v16bf bw[8];
#pragma unroll
      for (int pt = 0; pt < 8; ++pt)
        bw[pt] = load_bt16(Whh, Hn, (pt >> 1) * Hn + w * 32 + (pt & 1) * 16, kk * 32, lane);
#pragma unroll
      for (int pt = 0; pt < 8; ++pt)
        acc[pt] = WMMA_BF16(ah, bw[pt], acc[pt]);
    }
    __syncthreads();  // all h_bf reads done
#pragma unroll
    for (int t = 0; t < 2; ++t) {
      int col = w * 32 + t * 16 + n;
#pragma unroll
      for (int r = 0; r < 8; ++r) {
        int row = rbase + r;
        float gi = acc[0 + t][r], gf = acc[2 + t][r], gg = acc[4 + t][r], go = acc[6 + t][r];
        float c = sigm(gf) * creg[t][r] + sigm(gi) * tanhf(gg);
        float h = sigm(go) * tanhf(c);
        creg[t][r] = c;
        h_bf[row * Hn + col] = f2bf(h);
        hout[(size_t)l * Bn * Hn + row * Hn + col] = h;
      }
    }
    __syncthreads();  // new h_bf visible
  }
}

// ---------------- pack encoded [B][L][2H] (f32 + bf16) ----------------
__global__ __launch_bounds__(256)
void pack_encoded_k(const float* __restrict__ hF, const float* __restrict__ hB,
                    float* __restrict__ enc32, bf16_t* __restrict__ encbf) {
  size_t idx = (size_t)blockIdx.x * 256 + threadIdx.x;
  if (idx >= (size_t)Bn * Ln * H2) return;
  int j = (int)(idx & (H2 - 1));
  size_t bl = idx >> 9;            // b*Ln + l
  int l = (int)(bl % Ln);
  int b = (int)(bl / Ln);
  float v = (j < Hn) ? hF[((size_t)l * Bn + b) * Hn + j]
                     : hB[((size_t)l * Bn + b) * Hn + (j - Hn)];
  enc32[idx] = v;
  encbf[idx] = f2bf(v);
}

// ---------------- Wh_enc = encoded @ Wh^T + Wh_b : [6400][256] ----------------
__global__ __launch_bounds__(256)
void whenc_gemm_k(const bf16_t* __restrict__ enc, const bf16_t* __restrict__ Wh,
                  const float* __restrict__ Whb, float* __restrict__ out) {
  int tid = threadIdx.x, lane = tid & 31, w = tid >> 5;
  int mrow0 = blockIdx.x * 16;
  const bf16_t* A = enc + (size_t)mrow0 * H2;
  int n = lane & 15, rbase = (lane & 16) ? 8 : 0;
  v8f acc[2];
#pragma unroll
  for (int t = 0; t < 2; ++t) {
    float bv = Whb[w * 32 + t * 16 + n];
#pragma unroll
    for (int r = 0; r < 8; ++r) acc[t][r] = bv;
  }
#pragma unroll
  for (int kk = 0; kk < 16; ++kk) {
    v16bf a = load_a16(A, H2, kk * 32, lane);
    v16bf b0 = load_bt16(Wh, H2, w * 32, kk * 32, lane);
    v16bf b1 = load_bt16(Wh, H2, w * 32 + 16, kk * 32, lane);
    acc[0] = WMMA_BF16(a, b0, acc[0]);
    acc[1] = WMMA_BF16(a, b1, acc[1]);
  }
#pragma unroll
  for (int t = 0; t < 2; ++t)
#pragma unroll
    for (int r = 0; r < 8; ++r)
      out[(size_t)(mrow0 + rbase + r) * Hn + w * 32 + t * 16 + n] = acc[t][r];
}

// ---------------- decoder LSTM cell + Ws/Wc terms (one block) ----------------
__global__ __launch_bounds__(256)
void decoder_cell_k(const int* __restrict__ target, int tstep,
                    const float* __restrict__ embed,
                    const bf16_t* __restrict__ WihD, const bf16_t* __restrict__ WhhD,
                    const float* __restrict__ bD,
                    const bf16_t* __restrict__ Ws, const float* __restrict__ WsB,
                    const float* __restrict__ Wc, const float* __restrict__ WcB,
                    bf16_t* __restrict__ h_g, float* __restrict__ c_g,
                    const float* __restrict__ cov,
                    float* __restrict__ s_out, bf16_t* __restrict__ cat_bf) {
  __shared__ bf16_t x_bf[Bn * En];
  __shared__ bf16_t h_bf[Bn * Hn];
  __shared__ float  sterm[Bn * Hn];
  int tid = threadIdx.x, lane = tid & 31, w = tid >> 5;

  for (int i = tid; i < Bn * En; i += 256) {
    int b = i >> 7, e = i & (En - 1);
    int tok = target[b * Tn + tstep];
    x_bf[i] = f2bf(embed[(size_t)tok * En + e]);
  }
  for (int i = tid; i < Bn * Hn; i += 256) h_bf[i] = h_g[i];
  __syncthreads();

  int n = lane & 15, rbase = (lane & 16) ? 8 : 0;
  v8f acc[8];
#pragma unroll
  for (int pt = 0; pt < 8; ++pt) {
    float bv = bD[(pt >> 1) * Hn + w * 32 + (pt & 1) * 16 + n];
    v8f a;
#pragma unroll
    for (int r = 0; r < 8; ++r) a[r] = bv;
    acc[pt] = a;
  }
#pragma unroll
  for (int kk = 0; kk < 4; ++kk) {
    v16bf af = load_a16(x_bf, En, kk * 32, lane);
    v16bf bw[8];
#pragma unroll
    for (int pt = 0; pt < 8; ++pt)
      bw[pt] = load_bt16(WihD, En, (pt >> 1) * Hn + w * 32 + (pt & 1) * 16, kk * 32, lane);
#pragma unroll
    for (int pt = 0; pt < 8; ++pt)
      acc[pt] = WMMA_BF16(af, bw[pt], acc[pt]);
  }
#pragma unroll
  for (int kk = 0; kk < 8; ++kk) {
    v16bf ah = load_a16(h_bf, Hn, kk * 32, lane);
    v16bf bw[8];
#pragma unroll
    for (int pt = 0; pt < 8; ++pt)
      bw[pt] = load_bt16(WhhD, Hn, (pt >> 1) * Hn + w * 32 + (pt & 1) * 16, kk * 32, lane);
#pragma unroll
    for (int pt = 0; pt < 8; ++pt)
      acc[pt] = WMMA_BF16(ah, bw[pt], acc[pt]);
  }
  __syncthreads();  // done reading old h_bf
#pragma unroll
  for (int t = 0; t < 2; ++t) {
    int col = w * 32 + t * 16 + n;
#pragma unroll
    for (int r = 0; r < 8; ++r) {
      int row = rbase + r;
      float gi = acc[0 + t][r], gf = acc[2 + t][r], gg = acc[4 + t][r], go = acc[6 + t][r];
      float c = sigm(gf) * c_g[row * Hn + col] + sigm(gi) * tanhf(gg);
      float h = sigm(go) * tanhf(c);
      c_g[row * Hn + col] = c;
      bf16_t hb = f2bf(h);
      h_bf[row * Hn + col] = hb;
      h_g[row * Hn + col] = hb;
      cat_bf[row * H3 + col] = hb;
    }
  }
  __syncthreads();

  // ws_term = h @ Ws^T + Ws_b  (WMMA, K=256)
  v8f wacc[2];
#pragma unroll
  for (int t = 0; t < 2; ++t) {
    float bv = WsB[w * 32 + t * 16 + n];
#pragma unroll
    for (int r = 0; r < 8; ++r) wacc[t][r] = bv;
  }
#pragma unroll
  for (int kk = 0; kk < 8; ++kk) {
    v16bf ah = load_a16(h_bf, Hn, kk * 32, lane);
    v16bf b0 = load_bt16(Ws, Hn, w * 32, kk * 32, lane);
    v16bf b1 = load_bt16(Ws, Hn, w * 32 + 16, kk * 32, lane);
    wacc[0] = WMMA_BF16(ah, b0, wacc[0]);
    wacc[1] = WMMA_BF16(ah, b1, wacc[1]);
  }
#pragma unroll
  for (int t = 0; t < 2; ++t)
#pragma unroll
    for (int r = 0; r < 8; ++r)
      sterm[(rbase + r) * Hn + w * 32 + t * 16 + n] = wacc[t][r];
  __syncthreads();

  // wc_term (K=400, f32 VALU) ; s = ws + wc
#pragma unroll 1
  for (int e = 0; e < 16; ++e) {
    int q = tid + 256 * e;
    int b = q >> 8, g = q & (Hn - 1);
    const float* cv = cov + b * Ln;
    const float* wr = Wc + (size_t)g * Ln;
    float sum = sterm[b * Hn + g] + WcB[g];
    for (int ll = 0; ll < Ln; ++ll) sum = fmaf(cv[ll], wr[ll], sum);
    s_out[b * Hn + g] = sum;
  }
}

// ---------------- attention + coverage + context (grid = B) ----------------
__global__ __launch_bounds__(256)
void attention_step_k(const float* __restrict__ WhEnc, const float* __restrict__ s,
                      const float* __restrict__ vw, const float* __restrict__ vb,
                      const float* __restrict__ enc, float* __restrict__ cov,
                      float* __restrict__ covloss, bf16_t* __restrict__ cat_bf) {
  __shared__ float logi[Ln];
  __shared__ float red[256];
  int b = blockIdx.x, tid = threadIdx.x;

  for (int l = tid; l < Ln; l += 256) {
    int srow = (b * Ln + l) & (Bn - 1);   // faithful to .repeat tiling bug
    const float* wrow = WhEnc + (size_t)(b * Ln + l) * Hn;
    const float* sr = s + srow * Hn;
    float acc = vb[0];
    for (int hh = 0; hh < Hn; ++hh) acc = fmaf(vw[hh], wrow[hh] + sr[hh], acc);
    logi[l] = acc;
  }
  __syncthreads();
  float m = -1e30f;
  for (int l = tid; l < Ln; l += 256) m = fmaxf(m, logi[l]);
  red[tid] = m; __syncthreads();
  for (int st = 128; st > 0; st >>= 1) { if (tid < st) red[tid] = fmaxf(red[tid], red[tid + st]); __syncthreads(); }
  m = red[0]; __syncthreads();
  float sm = 0.0f;
  for (int l = tid; l < Ln; l += 256) { float e = __expf(logi[l] - m); logi[l] = e; sm += e; }
  red[tid] = sm; __syncthreads();
  for (int st = 128; st > 0; st >>= 1) { if (tid < st) red[tid] += red[tid + st]; __syncthreads(); }
  float inv = 1.0f / red[0]; __syncthreads();
  float mloc = 0.0f;
  for (int l = tid; l < Ln; l += 256) {
    float a = logi[l] * inv; logi[l] = a;
    float cv = cov[b * Ln + l];
    mloc += fminf(a, cv);
    cov[b * Ln + l] = cv + a;
  }
  red[tid] = mloc; __syncthreads();
  for (int st = 128; st > 0; st >>= 1) { if (tid < st) red[tid] += red[tid + st]; __syncthreads(); }
  if (tid == 0) atomicAdd(covloss, red[0]);
  __syncthreads();
  for (int j = tid; j < H2; j += 256) {
    const float* e = enc + (size_t)b * Ln * H2 + j;
    float cacc = 0.0f;
    for (int l = 0; l < Ln; ++l) cacc = fmaf(logi[l], e[(size_t)l * H2], cacc);
    cat_bf[b * H3 + Hn + j] = f2bf(cacc);
  }
}

// ---------------- fc1: cat @ V1^T + V1_b -> [16][768] bf16 ----------------
__global__ __launch_bounds__(256)
void fc1_step_k(const bf16_t* __restrict__ cat, const bf16_t* __restrict__ V1,
                const float* __restrict__ V1b, bf16_t* __restrict__ out_bf) {
  int tid = threadIdx.x, lane = tid & 31, w = tid >> 5;
  int n = lane & 15, rbase = (lane & 16) ? 8 : 0;
  v8f acc[6];
#pragma unroll
  for (int t = 0; t < 6; ++t) {
    float bv = V1b[w * 96 + t * 16 + n];
#pragma unroll
    for (int r = 0; r < 8; ++r) acc[t][r] = bv;
  }
#pragma unroll
  for (int kk = 0; kk < 24; ++kk) {
    v16bf a = load_a16(cat, H3, kk * 32, lane);
    v16bf bw[6];
#pragma unroll
    for (int t = 0; t < 6; ++t)
      bw[t] = load_bt16(V1, H3, w * 96 + t * 16, kk * 32, lane);
#pragma unroll
    for (int t = 0; t < 6; ++t)
      acc[t] = WMMA_BF16(a, bw[t], acc[t]);
  }
#pragma unroll
  for (int t = 0; t < 6; ++t)
#pragma unroll
    for (int r = 0; r < 8; ++r)
      out_bf[(rbase + r) * H3 + w * 96 + t * 16 + n] = f2bf(acc[t][r]);
}

// ---------------- fc2: fc1 @ V2^T + V2_b -> logits in d_out ----------------
// The activations [16][768] are staged in LDS once per block; each wave then
// streams only its V2 B-fragments from the L2-resident bf16 weight copy.
__global__ __launch_bounds__(256)
void fc2_step_k(const bf16_t* __restrict__ fin, const bf16_t* __restrict__ V2,
                const float* __restrict__ V2b, float* __restrict__ dout, int tstep) {
  __shared__ bf16_t fin_s[Bn * H3];
  int tid = threadIdx.x, lane = tid & 31, w = tid >> 5;
  for (int i = tid; i < Bn * H3; i += 256) fin_s[i] = fin[i];
  __syncthreads();

  int tile = blockIdx.x * 8 + w;
  if (tile >= Vn / 16) return;   // wave-uniform, after the barrier
  int n0 = tile * 16;
  int n = lane & 15, rbase = (lane & 16) ? 8 : 0;
  v8f acc;
  float bv = V2b[n0 + n];
#pragma unroll
  for (int r = 0; r < 8; ++r) acc[r] = bv;
#pragma unroll
  for (int kk = 0; kk < 24; ++kk) {
    v16bf a = load_a16(fin_s, H3, kk * 32, lane);
    acc = WMMA_BF16(a, load_bt16(V2, H3, n0, kk * 32, lane), acc);
  }
#pragma unroll
  for (int r = 0; r < 8; ++r) {
    int b = rbase + r;
    dout[((size_t)b * NT + tstep) * Vn + n0 + n] = acc[r];
  }
}

// ---------------- softmax over V, in place in d_out (grid = B) ----------------
__global__ __launch_bounds__(256)
void softmax_vocab_k(float* __restrict__ dout, int tstep) {
  int b = blockIdx.x, tid = threadIdx.x;
  float* row = dout + ((size_t)b * NT + tstep) * Vn;
  __shared__ float red[256];
  float m = -1e30f;
  for (int v = tid; v < Vn; v += 256) m = fmaxf(m, row[v]);
  red[tid] = m; __syncthreads();
  for (int st = 128; st > 0; st >>= 1) { if (tid < st) red[tid] = fmaxf(red[tid], red[tid + st]); __syncthreads(); }
  m = red[0]; __syncthreads();
  float s = 0.0f;
  for (int v = tid; v < Vn; v += 256) s += __expf(row[v] - m);
  red[tid] = s; __syncthreads();
  for (int st = 128; st > 0; st >>= 1) { if (tid < st) red[tid] += red[tid + st]; __syncthreads(); }
  float inv = 1.0f / red[0];
  for (int v = tid; v < Vn; v += 256) row[v] = __expf(row[v] - m) * inv;
}

__global__ void write_covloss_k(const float* __restrict__ covloss, float* __restrict__ dout) {
  if (threadIdx.x == 0 && blockIdx.x == 0)
    dout[(size_t)Bn * NT * Vn] = covloss[0];
}

// ---------------- host launcher ----------------
extern "C" void kernel_launch(void* const* d_in, const int* in_sizes, int n_in,
                              void* d_out, int out_size, void* d_ws, size_t ws_size,
                              hipStream_t stream) {
  (void)in_sizes; (void)n_in; (void)out_size; (void)ws_size;
  const int*   inputs = (const int*)d_in[0];
  const int*   target = (const int*)d_in[1];
  const float* embed  = (const float*)d_in[2];
  const float* Wih_f = (const float*)d_in[3];
  const float* Whh_f = (const float*)d_in[4];
  const float* b_f   = (const float*)d_in[5];
  const float* Wih_b = (const float*)d_in[6];
  const float* Whh_b = (const float*)d_in[7];
  const float* b_b   = (const float*)d_in[8];
  const float* Wih_d = (const float*)d_in[9];
  const float* Whh_d = (const float*)d_in[10];
  const float* b_d   = (const float*)d_in[11];
  const float* Wh_w  = (const float*)d_in[12];
  const float* Wh_b  = (const float*)d_in[13];
  const float* Ws_w  = (const float*)d_in[14];
  const float* Ws_b  = (const float*)d_in[15];
  const float* Wc_w  = (const float*)d_in[16];
  const float* Wc_b  = (const float*)d_in[17];
  const float* v_w   = (const float*)d_in[18];
  const float* v_b   = (const float*)d_in[19];
  const float* V1_w  = (const float*)d_in[20];
  const float* V1_b  = (const float*)d_in[21];
  const float* V2_w  = (const float*)d_in[22];
  const float* V2_b  = (const float*)d_in[23];
  float* dout = (float*)d_out;

  char* base = (char*)d_ws;
  size_t off = 0;
  auto alloc = [&](size_t bytes) -> char* {
    char* p = base + off;
    off = (off + bytes + 255) & ~(size_t)255;
    return p;
  };
  bf16_t* wWihF = (bf16_t*)alloc((size_t)G4 * En * 2);
  bf16_t* wWhhF = (bf16_t*)alloc((size_t)G4 * Hn * 2);
  bf16_t* wWihB = (bf16_t*)alloc((size_t)G4 * En * 2);
  bf16_t* wWhhB = (bf16_t*)alloc((size_t)G4 * Hn * 2);
  bf16_t* wWihD = (bf16_t*)alloc((size_t)G4 * En * 2);
  bf16_t* wWhhD = (bf16_t*)alloc((size_t)G4 * Hn * 2);
  bf16_t* wWh   = (bf16_t*)alloc((size_t)Hn * H2 * 2);
  bf16_t* wWs   = (bf16_t*)alloc((size_t)Hn * Hn * 2);
  bf16_t* wV1   = (bf16_t*)alloc((size_t)H3 * H3 * 2);
  bf16_t* wV2   = (bf16_t*)alloc((size_t)Vn * H3 * 2);
  bf16_t* xsb   = (bf16_t*)alloc((size_t)Ln * Bn * En * 2);
  float*  hF    = (float*)alloc((size_t)Ln * Bn * Hn * 4);
  float*  hB    = (float*)alloc((size_t)Ln * Bn * Hn * 4);
  float*  enc32 = (float*)alloc((size_t)Bn * Ln * H2 * 4);
  bf16_t* encbf = (bf16_t*)alloc((size_t)Bn * Ln * H2 * 2);
  float*  whenc = (float*)alloc((size_t)Bn * Ln * Hn * 4);
  float*  cov   = (float*)alloc((size_t)Bn * Ln * 4);
  float*  c_g   = (float*)alloc((size_t)Bn * Hn * 4);
  bf16_t* h_g   = (bf16_t*)alloc((size_t)Bn * Hn * 2);
  float*  s_t   = (float*)alloc((size_t)Bn * Hn * 4);
  bf16_t* catb  = (bf16_t*)alloc((size_t)Bn * H3 * 2);
  bf16_t* fc1b  = (bf16_t*)alloc((size_t)Bn * H3 * 2);
  float*  cls   = (float*)alloc(4);

  auto cvt = [&](const float* src, bf16_t* dst, size_t n) {
    f32_to_bf16_k<<<(unsigned)((n + 255) / 256), 256, 0, stream>>>(src, dst, n);
  };
  cvt(Wih_f, wWihF, (size_t)G4 * En);
  cvt(Whh_f, wWhhF, (size_t)G4 * Hn);
  cvt(Wih_b, wWihB, (size_t)G4 * En);
  cvt(Whh_b, wWhhB, (size_t)G4 * Hn);
  cvt(Wih_d, wWihD, (size_t)G4 * En);
  cvt(Whh_d, wWhhD, (size_t)G4 * Hn);
  cvt(Wh_w,  wWh,   (size_t)Hn * H2);
  cvt(Ws_w,  wWs,   (size_t)Hn * Hn);
  cvt(V1_w,  wV1,   (size_t)H3 * H3);
  cvt(V2_w,  wV2,   (size_t)Vn * H3);

  gather_xs_k<<<(Ln * Bn * En + 255) / 256, 256, 0, stream>>>(inputs, embed, xsb);
  init_state_k<<<(Bn * Ln + 255) / 256, 256, 0, stream>>>(cov, c_g, h_g, cls);

  lstm_encoder_scan_k<<<2, 256, 0, stream>>>(xsb, wWihF, wWhhF, b_f,
                                             wWihB, wWhhB, b_b, hF, hB);

  pack_encoded_k<<<(Bn * Ln * H2 + 255) / 256, 256, 0, stream>>>(hF, hB, enc32, encbf);
  whenc_gemm_k<<<Bn * Ln / 16, 256, 0, stream>>>(encbf, wWh, Wh_b, whenc);

  for (int t = 0; t < NT; ++t) {
    decoder_cell_k<<<1, 256, 0, stream>>>(target, t, embed, wWihD, wWhhD, b_d,
                                          wWs, Ws_b, Wc_w, Wc_b,
                                          h_g, c_g, cov, s_t, catb);
    attention_step_k<<<Bn, 256, 0, stream>>>(whenc, s_t, v_w, v_b, enc32,
                                             cov, cls, catb);
    fc1_step_k<<<1, 256, 0, stream>>>(catb, wV1, V1_b, fc1b);
    fc2_step_k<<<(Vn / 16 + 7) / 8, 256, 0, stream>>>(fc1b, wV2, V2_b, dout, t);
    softmax_vocab_k<<<Bn, 256, 0, stream>>>(dout, t);
  }
  write_covloss_k<<<1, 1, 0, stream>>>(cls, dout);
}
